// NeuralPlannerGAT_49400713838997
// MI455X (gfx1250) — matrix operand order repaired and checked by reading, not compiled, not run on hardware
//
#include <hip/hip_runtime.h>

#define NEG_SLOPE 0.2f
#define BN_EPS    1e-5f

typedef __attribute__((ext_vector_type(16))) __bf16 v16bf;
typedef __attribute__((ext_vector_type(8)))  float  v8f;
typedef __attribute__((ext_vector_type(4)))  int    v4i;

// All WMMA GEMMs in this net have K = 256.
#define GEMM_K  256
#define GEMM_KT (GEMM_K / 32)          // 8 K-tiles of 32
#define NT_PER_BLOCK 4                 // 4 N-tiles (64 cols) per block
#define LDSB_ELEMS (NT_PER_BLOCK * GEMM_KT * 512)   // 16384 bf16 = 32 KB

// ---------- order-preserving float<->unsigned encoding for atomicMax ----------
static __device__ __forceinline__ unsigned enc_f32(float f) {
  unsigned u = __float_as_uint(f);
  return (u & 0x80000000u) ? ~u : (u | 0x80000000u);
}
static __device__ __forceinline__ float dec_f32(unsigned k) {
  unsigned u = (k & 0x80000000u) ? (k ^ 0x80000000u) : ~k;
  return __uint_as_float(u);
}

// ---------- convert + permute weights into WMMA B-fragment order ----------
// Packed layout: idx = (((t*KT + q)*32) + lane)*16 + j
//   maps to W[k = 32q + (lane>>4)*16 + j , n = 16t + (lane&15)]
__global__ void k_pack_b(const float* __restrict__ W, __bf16* __restrict__ out,
                         int Nc) {
  int i = blockIdx.x * 256 + threadIdx.x;
  int total = GEMM_K * Nc;
  if (i >= total) return;
  int j  = i & 15;
  int l  = (i >> 4) & 31;
  int qq = i >> 9;                 // t*KT + q
  int t  = qq >> 3;                // KT == 8
  int q  = qq & 7;
  int k  = q * 32 + (l >> 4) * 16 + j;
  int n  = t * 16 + (l & 15);
  out[i] = (__bf16)W[k * Nc + n];
}

// ---------- layer-1 linear, K = FIN = 5 (too small for WMMA) ----------
__global__ void k_lin_fin(const float* __restrict__ x, const float* __restrict__ W,
                          const float* __restrict__ bias, float* __restrict__ out,
                          int M, int Nc) {
  int i = blockIdx.x * 256 + threadIdx.x;
  if (i >= M * Nc) return;
  int n = i % Nc, m = i / Nc;
  const float* xp = x + (size_t)m * 5;
  float acc = bias[n];
#pragma unroll
  for (int k = 0; k < 5; ++k) acc += xp[k] * W[k * Nc + n];
  out[i] = acc;
}

// ---------- bf16 WMMA GEMM: C[M,Nc] = A[M,256] * B[256,Nc] (+bias) ----------
// 4 waves/block. Block stages packed-B for its 4 N-tiles into LDS (async path),
// each wave computes one 16-row M tile across those 4 N-tiles (32 WMMAs).
__global__ __launch_bounds__(128) void k_gemm_bf16(
    const __bf16* __restrict__ A, const __bf16* __restrict__ Bpack,
    const float* __restrict__ bias, float* __restrict__ C,
    int M, int Nc) {
  __shared__ __align__(32) __bf16 lB[LDSB_ELEMS];

  const int tid  = threadIdx.x;
  const int wave = tid >> 5;
  const int lane = tid & 31;
  const int mtile = blockIdx.x * 4 + wave;
  const int m0 = mtile * 16;
  const int nt0 = blockIdx.y * NT_PER_BLOCK;
  const int row  = lane & 15;
  const int half = lane >> 4;

  // ---- stage this block's packed-B chunk (contiguous 32 KB) into LDS ----
  {
    const __bf16* srcB = Bpack + (size_t)nt0 * GEMM_KT * 512;
    const int nVec = LDSB_ELEMS / 8;          // 16-byte chunks
#if __has_builtin(__builtin_amdgcn_global_load_async_to_lds_b128)
    v4i* g = (v4i*)srcB;
    v4i* l = (v4i*)lB;
    for (int v = tid; v < nVec; v += 128)
      __builtin_amdgcn_global_load_async_to_lds_b128(g + v, l + v, 0, 0);
    asm volatile("s_wait_asynccnt 0x0" ::: "memory");
#else
    const int4* s4 = (const int4*)srcB;
    int4* d4 = (int4*)lB;
    for (int v = tid; v < nVec; v += 128) d4[v] = s4[v];
#endif
    __syncthreads();
  }

  if (m0 >= M) return;                        // wave-uniform: EXEC all-1s for WMMA

  int ar = m0 + row; if (ar > M - 1) ar = M - 1;    // clamp keeps EXEC full
  const __bf16* aRow = A + (size_t)ar * GEMM_K;

  v8f acc[NT_PER_BLOCK] = {};
#pragma unroll
  for (int q = 0; q < GEMM_KT; ++q) {
    v16bf a;
    const __bf16* ap = aRow + q * 32 + half * 16;
#pragma unroll
    for (int j = 0; j < 8; ++j) { a[j] = ap[j]; a[8 + j] = ap[8 + j]; }
#pragma unroll
    for (int t = 0; t < NT_PER_BLOCK; ++t) {
      const v16bf b = *(const v16bf*)&lB[((t * GEMM_KT + q) * 32 + lane) * 16];
      acc[t] = __builtin_amdgcn_wmma_f32_16x16x32_bf16(false, a, false, b,
                                                       (short)0, acc[t],
                                                       false, false);
    }
  }

#pragma unroll
  for (int t = 0; t < NT_PER_BLOCK; ++t) {
    const int nc = (nt0 + t) * 16 + row;
    const float bv = bias ? bias[nc] : 0.0f;
#pragma unroll
    for (int v = 0; v < 8; ++v) {
      int mr = m0 + half * 8 + v;             // C layout: VGPR v -> row half*8+v
      if (mr < M) C[(size_t)mr * Nc + nc] = acc[t][v] + bv;
    }
  }
}

// ---------- per-layer scratch init ----------
__global__ void k_init(float* __restrict__ agg, unsigned* __restrict__ amax,
                       float* __restrict__ den, int ND, int NH) {
  int i = blockIdx.x * 256 + threadIdx.x;
  if (i < ND) agg[i] = 0.0f;
  if (i < NH) { amax[i] = 0u; den[i] = 0.0f; }   // enc 0 == -inf sentinel
}

// ---------- edge pass 1: attention logits + segment max ----------
__global__ void k_edge_score(const float* __restrict__ xl, const float* __restrict__ xr,
                             const int* __restrict__ src, const int* __restrict__ dst,
                             const float* __restrict__ att, float* __restrict__ a,
                             unsigned* __restrict__ amax, int E, int H) {
  int i = blockIdx.x * 256 + threadIdx.x;
  if (i >= E * H) return;
  int e = i / H, h = i - e * H;
  int s = src[e], d = dst[e];
  const float* pl = xl + ((size_t)s * H + h) * 64;
  const float* pr = xr + ((size_t)d * H + h) * 64;
  const float* pa = att + h * 64;
  float acc = 0.0f;
#pragma unroll 8
  for (int c = 0; c < 64; ++c) {
    float u = pl[c] + pr[c];
    u = u > 0.0f ? u : NEG_SLOPE * u;            // leaky_relu
    acc += u * pa[c];
  }
  a[i] = acc;
  atomicMax(&amax[d * H + h], enc_f32(acc));
}

// ---------- edge pass 2: exp(a - max) + segment sum ----------
__global__ void k_edge_exp(const float* __restrict__ a, const int* __restrict__ dst,
                           const unsigned* __restrict__ amax, float* __restrict__ ex,
                           float* __restrict__ den, int E, int H) {
  int i = blockIdx.x * 256 + threadIdx.x;
  if (i >= E * H) return;
  int e = i / H, h = i - e * H;
  int d = dst[e];
  float v = __expf(a[i] - dec_f32(amax[d * H + h]));
  ex[i] = v;
  atomicAdd(&den[d * H + h], v);
}

// ---------- edge pass 3: alpha-weighted scatter of xl[src] ----------
__global__ void k_edge_agg(const float* __restrict__ xl, const int* __restrict__ src,
                           const int* __restrict__ dst, const float* __restrict__ ex,
                           const float* __restrict__ den, float* __restrict__ agg,
                           int E, int H, int D) {
  long long i = (long long)blockIdx.x * 256 + threadIdx.x;
  long long total = (long long)E * D;
  if (i >= total) return;
  int e = (int)(i / D);
  int dch = (int)(i - (long long)e * D);
  int h = dch >> 6;                              // 64 channels per head
  int s = src[e], d = dst[e];
  float alpha = ex[e * H + h] / (den[d * H + h] + 1e-16f);
  atomicAdd(&agg[(size_t)d * D + dch], alpha * xl[(size_t)s * D + dch]);
}

// ---------- bias (+res) + BatchNorm + ELU, plus bf16 shadow copy ----------
__global__ void k_node_post(const float* __restrict__ agg, const float* __restrict__ res,
                            const float* __restrict__ cb,
                            const float* __restrict__ g, const float* __restrict__ b,
                            const float* __restrict__ m, const float* __restrict__ v,
                            float* __restrict__ h, __bf16* __restrict__ hbf,
                            int total, int D) {
  int i = blockIdx.x * 256 + threadIdx.x;
  if (i >= total) return;
  int dch = i % D;
  float val = agg[i] + cb[dch];
  if (res) val += res[i];
  val = (val - m[dch]) * rsqrtf(v[dch] + BN_EPS) * g[dch] + b[dch];
  val = val > 0.0f ? val : (__expf(val) - 1.0f);  // ELU
  h[i] = val;
  if (hbf) hbf[i] = (__bf16)val;
}

// ---------- final MLP: concat([h3, x]) -> 64 -> 32 -> 1, one block per node ----------
__global__ __launch_bounds__(64) void k_mlp(const float* __restrict__ h3,
                                            const float* __restrict__ x,
                                            const float* __restrict__ W1, const float* __restrict__ b1,
                                            const float* __restrict__ W2, const float* __restrict__ b2,
                                            const float* __restrict__ W3, const float* __restrict__ b3,
                                            float* __restrict__ out, int Ntot) {
  __shared__ float z1[64];
  __shared__ float z2[32];
  int n = blockIdx.x;
  if (n >= Ntot) return;
  int t = threadIdx.x;
  const float* hp = h3 + (size_t)n * 128;
  const float* xp = x + (size_t)n * 5;
  float acc = b1[t];
  for (int k = 0; k < 128; ++k) acc += hp[k] * W1[k * 64 + t];
#pragma unroll
  for (int k = 0; k < 5; ++k) acc += xp[k] * W1[(128 + k) * 64 + t];
  z1[t] = fmaxf(acc, 0.0f);
  __syncthreads();
  if (t < 32) {
    float a2 = b2[t];
    for (int k = 0; k < 64; ++k) a2 += z1[k] * W2[k * 32 + t];
    z2[t] = fmaxf(a2, 0.0f);
  }
  __syncthreads();
  if (t == 0) {
    float a3 = b3[0];
#pragma unroll
    for (int k = 0; k < 32; ++k) a3 += z2[k] * W3[k];
    out[n] = a3;
  }
}

extern "C" void kernel_launch(void* const* d_in, const int* in_sizes, int n_in,
                              void* d_out, int out_size, void* d_ws, size_t ws_size,
                              hipStream_t stream) {
  (void)n_in; (void)out_size; (void)ws_size;
  const float* x  = (const float*)d_in[0];
  const int*   ei = (const int*)d_in[1];
  const int N = in_sizes[0] / 5;
  const int E = in_sizes[1] / 2;
  const int* src = ei;
  const int* dst = ei + E;

  const float *c1_Wl = (const float*)d_in[2],  *c1_bl = (const float*)d_in[3];
  const float *c1_Wr = (const float*)d_in[4],  *c1_br = (const float*)d_in[5];
  const float *c1_att = (const float*)d_in[6], *c1_b = (const float*)d_in[7];
  const float *bn1_g = (const float*)d_in[8],  *bn1_b = (const float*)d_in[9];
  const float *bn1_m = (const float*)d_in[10], *bn1_v = (const float*)d_in[11];
  const float *bn2_g = (const float*)d_in[12], *bn2_b = (const float*)d_in[13];
  const float *bn2_m = (const float*)d_in[14], *bn2_v = (const float*)d_in[15];
  const float *bn3_g = (const float*)d_in[16], *bn3_b = (const float*)d_in[17];
  const float *bn3_m = (const float*)d_in[18], *bn3_v = (const float*)d_in[19];
  const float *c2_Wl = (const float*)d_in[20], *c2_bl = (const float*)d_in[21];
  const float *c2_Wr = (const float*)d_in[22], *c2_br = (const float*)d_in[23];
  const float *c2_att = (const float*)d_in[24], *c2_b = (const float*)d_in[25];
  const float *c2_Wres = (const float*)d_in[26];
  const float *c3_Wl = (const float*)d_in[27], *c3_bl = (const float*)d_in[28];
  const float *c3_Wr = (const float*)d_in[29], *c3_br = (const float*)d_in[30];
  const float *c3_att = (const float*)d_in[31], *c3_b = (const float*)d_in[32];
  const float *c3_Wres = (const float*)d_in[33];
  const float *W1 = (const float*)d_in[34], *b1 = (const float*)d_in[35];
  const float *W2 = (const float*)d_in[36], *b2 = (const float*)d_in[37];
  const float *W3 = (const float*)d_in[38], *b3 = (const float*)d_in[39];

  // ---- bump allocator over d_ws ----
  char* p = (char*)d_ws;
  auto alloc = [&](size_t bytes) -> void* {
    void* r = (void*)p;
    p += (bytes + 255) & ~(size_t)255;
    return r;
  };
  float*    xl   = (float*)alloc((size_t)N * 256 * 4);
  float*    xr   = (float*)alloc((size_t)N * 256 * 4);
  float*    res  = (float*)alloc((size_t)N * 256 * 4);
  float*    agg  = (float*)alloc((size_t)N * 256 * 4);
  float*    h    = (float*)alloc((size_t)N * 256 * 4);
  __bf16*   hbf  = (__bf16*)alloc((size_t)N * 256 * 2);
  __bf16*   wbfL = (__bf16*)alloc((size_t)256 * 256 * 2);
  __bf16*   wbfR = (__bf16*)alloc((size_t)256 * 256 * 2);
  __bf16*   wbfS = (__bf16*)alloc((size_t)256 * 256 * 2);
  float*    ae   = (float*)alloc((size_t)E * 4 * 4);
  float*    exb  = (float*)alloc((size_t)E * 4 * 4);
  unsigned* amax = (unsigned*)alloc((size_t)N * 4 * 4);
  float*    den  = (float*)alloc((size_t)N * 4 * 4);

  auto cdiv = [](long long a, long long b) { return (unsigned)((a + b - 1) / b); };

  auto gemm = [&](const __bf16* A, const __bf16* Bpack, const float* bias, float* C,
                  int M, int Nc) {
    dim3 grid(cdiv(cdiv(M, 16), 4), Nc / (16 * NT_PER_BLOCK));
    k_gemm_bf16<<<grid, 128, 0, stream>>>(A, Bpack, bias, C, M, Nc);
  };
  auto run_edges = [&](const float* att, int H, int D) {
    int ND = N * D, NH = N * H, EH = E * H;
    long long ED = (long long)E * D;
    k_init<<<cdiv(ND, 256), 256, 0, stream>>>(agg, amax, den, ND, NH);
    k_edge_score<<<cdiv(EH, 256), 256, 0, stream>>>(xl, xr, src, dst, att, ae, amax, E, H);
    k_edge_exp<<<cdiv(EH, 256), 256, 0, stream>>>(ae, dst, amax, exb, den, E, H);
    k_edge_agg<<<cdiv(ED, 256), 256, 0, stream>>>(xl, src, dst, exb, den, agg, E, H, D);
  };

  const int T1 = N * 256, T3 = N * 128;

  // ---------------- Layer 1 (FIN=5 -> 256, 4 heads x 64) ----------------
  k_lin_fin<<<cdiv(T1, 256), 256, 0, stream>>>(x, c1_Wl, c1_bl, xl, N, 256);
  k_lin_fin<<<cdiv(T1, 256), 256, 0, stream>>>(x, c1_Wr, c1_br, xr, N, 256);
  run_edges(c1_att, 4, 256);
  k_node_post<<<cdiv(T1, 256), 256, 0, stream>>>(agg, (const float*)nullptr, c1_b,
      bn1_g, bn1_b, bn1_m, bn1_v, h, hbf, T1, 256);

  // ---------------- Layer 2 (256 -> 256, 4 heads x 64, +res) ----------------
  k_pack_b<<<cdiv(65536, 256), 256, 0, stream>>>(c2_Wl, wbfL, 256);
  k_pack_b<<<cdiv(65536, 256), 256, 0, stream>>>(c2_Wr, wbfR, 256);
  k_pack_b<<<cdiv(65536, 256), 256, 0, stream>>>(c2_Wres, wbfS, 256);
  gemm(hbf, wbfL, c2_bl, xl, N, 256);
  gemm(hbf, wbfR, c2_br, xr, N, 256);
  gemm(hbf, wbfS, (const float*)nullptr, res, N, 256);
  run_edges(c2_att, 4, 256);
  k_node_post<<<cdiv(T1, 256), 256, 0, stream>>>(agg, res, c2_b,
      bn2_g, bn2_b, bn2_m, bn2_v, h, hbf, T1, 256);

  // ---------------- Layer 3 (256 -> 128, 2 heads x 64, +res) ----------------
  k_pack_b<<<cdiv(32768, 256), 256, 0, stream>>>(c3_Wl, wbfL, 128);
  k_pack_b<<<cdiv(32768, 256), 256, 0, stream>>>(c3_Wr, wbfR, 128);
  k_pack_b<<<cdiv(32768, 256), 256, 0, stream>>>(c3_Wres, wbfS, 128);
  gemm(hbf, wbfL, c3_bl, xl, N, 128);
  gemm(hbf, wbfR, c3_br, xr, N, 128);
  gemm(hbf, wbfS, (const float*)nullptr, res, N, 128);
  run_edges(c3_att, 2, 128);
  k_node_post<<<cdiv(T3, 256), 256, 0, stream>>>(agg, res, c3_b,
      bn3_g, bn3_b, bn3_m, bn3_v, h, (__bf16*)nullptr, T3, 128);

  // ---------------- MLP head ----------------
  k_mlp<<<N, 64, 0, stream>>>(h, x, W1, b1, W2, b2, W3, b3, (float*)d_out, N);
}
